// FusedThorExpertBmm_56788057588422
// MI455X (gfx1250) — compile-verified, matrix operand order checked by reading, not run on hardware
//
#include <hip/hip_runtime.h>
#include <hip/hip_bf16.h>

// FusedThorExpertBmm for MI455X (gfx1250, wave32, WMMA).
// Pass 1: h = gelu_exact(x @ w1 + b1)  -> bf16 in d_ws (needs E*C*I*2 = 128 MiB)
// Pass 2: y = h @ w2 + b2              -> fp32 d_out
// 128x128 block tiles, K=32 steps, v_wmma_f32_16x16x32_bf16,
// double-buffered LDS + software-pipelined global loads,
// async global->LDS copies for the bf16 A tile in pass 2.

#define NE 16
#define NC 1024
#define NH 1024
#define NI 4096

#define BM 128
#define BN 128
#define BK 32
#define LDP 40   // padded LDS k-stride (elements); 80 B keeps 16B align, spreads banks

typedef __attribute__((ext_vector_type(16))) __bf16 bf16x16;
typedef __attribute__((ext_vector_type(8)))  __bf16 bf16x8;
typedef __attribute__((ext_vector_type(8)))  float  f32x8;

union Frag { bf16x16 v; bf16x8 h[2]; };

__device__ __forceinline__ unsigned short f2bf(float f) {
  unsigned u = __float_as_uint(f);
  u += 0x7FFFu + ((u >> 16) & 1u);           // round-to-nearest-even
  return (unsigned short)(u >> 16);
}

// pack two floats into two bf16 (lo = a, hi = b)
__device__ __forceinline__ unsigned packbf2(float a, float b) {
#if __has_builtin(__builtin_amdgcn_cvt_pk_bf16_f32)
  auto p = __builtin_amdgcn_cvt_pk_bf16_f32(a, b);   // v_cvt_pk_bf16_f32
  return __builtin_bit_cast(unsigned, p);
#else
  return (unsigned)f2bf(a) | ((unsigned)f2bf(b) << 16);
#endif
}

#if __has_builtin(__builtin_amdgcn_global_load_async_to_lds_b128)
#define HAVE_ASYNC_COPY 1
typedef int v4i __attribute__((vector_size(16)));
typedef v4i __attribute__((address_space(1))) gv4i;   // global 16B chunk
typedef v4i __attribute__((address_space(3))) lv4i;   // LDS 16B chunk
__device__ __forceinline__ void wait_async0() {
#if __has_builtin(__builtin_amdgcn_s_wait_asynccnt)
  __builtin_amdgcn_s_wait_asynccnt(0);
#else
  asm volatile("s_wait_asynccnt 0x0" ::: "memory");
#endif
}
#endif

// ---------------------------------------------------------------- pass 1 ----
__global__ __launch_bounds__(256)
void thor_gemm1_gelu(const float* __restrict__ x,   // [E*C, H]
                     const float* __restrict__ w1,  // [E, H, I]
                     const float* __restrict__ b1,  // [E, 1, I]
                     unsigned short* __restrict__ hout) // [E, C, I] bf16
{
  __shared__ unsigned short ldsA[2][BM * LDP];   // x tile, row-major [m][k]
  __shared__ unsigned short ldsB[2][BN * LDP];   // w1 tile, transposed [n][k]

  const int e  = blockIdx.z;
  const int m0 = blockIdx.y * BM;
  const int n0 = blockIdx.x * BN;

  const int tid  = threadIdx.x;
  const int lane = tid & 31;
  const int wave = tid >> 5;
  const int wm   = wave & 1;      // 2 waves along M (64 rows each)
  const int wn   = wave >> 1;     // 4 waves along N (32 cols each)
  const int half = lane >> 4;     // K-half select per ISA A/B layout
  const int r    = lane & 15;

  const float* xe  = x  + (size_t)(e * NC + m0) * NH;
  const float* w1e = w1 + (size_t)e * NH * NI + n0;
  const float* b1e = b1 + (size_t)e * NI + n0;

  f32x8 acc[4][2];
  for (int i = 0; i < 4; ++i)
    for (int j = 0; j < 2; ++j)
      for (int t = 0; t < 8; ++t) acc[i][j][t] = 0.0f;

  float bias[2];
  bias[0] = b1e[wn * 32 + 0 * 16 + r];
  bias[1] = b1e[wn * 32 + 1 * 16 + r];

  float4 pa[4];        // prefetched A (128x32 fp32: 4 float4/thread)
  float4 pb[2][2];     // prefetched B (paired k-rows: 2 tasks x 2 rows)

  auto loadA = [&](int kb) {
    #pragma unroll
    for (int it = 0; it < 4; ++it) {
      int idx = tid + it * 256;               // 1024 float4 groups
      int row = idx >> 3;                     // 8 float4 per 32-wide row
      int c4  = (idx & 7) << 2;
      pa[it] = *(const float4*)(xe + (size_t)row * NH + kb + c4);
    }
  };
  auto storeA = [&](int buf) {
    #pragma unroll
    for (int it = 0; it < 4; ++it) {
      int idx = tid + it * 256;
      int row = idx >> 3;
      int c4  = (idx & 7) << 2;
      uint2 d;
      d.x = packbf2(pa[it].x, pa[it].y);
      d.y = packbf2(pa[it].z, pa[it].w);
      *(uint2*)&ldsA[buf][row * LDP + c4] = d;
    }
  };
  auto loadB = [&](int kb) {
    #pragma unroll
    for (int it = 0; it < 2; ++it) {
      int idx = tid + it * 256;               // 512 dual-row tasks
      int krp = idx >> 5;                     // k-pair 0..15
      int c4  = (idx & 31) << 2;              // col group (coalesced)
      const float* g = w1e + (size_t)(kb + 2 * krp) * NI + c4;
      pb[it][0] = *(const float4*)g;
      pb[it][1] = *(const float4*)(g + NI);
    }
  };
  auto storeB = [&](int buf) {                // transpose to [n][k], packed b32
    #pragma unroll
    for (int it = 0; it < 2; ++it) {
      int idx = tid + it * 256;
      int krp = idx >> 5;
      int c4  = (idx & 31) << 2;
      unsigned short* base = &ldsB[buf][0];
      *(unsigned*)&base[(c4 + 0) * LDP + 2 * krp] = packbf2(pb[it][0].x, pb[it][1].x);
      *(unsigned*)&base[(c4 + 1) * LDP + 2 * krp] = packbf2(pb[it][0].y, pb[it][1].y);
      *(unsigned*)&base[(c4 + 2) * LDP + 2 * krp] = packbf2(pb[it][0].z, pb[it][1].z);
      *(unsigned*)&base[(c4 + 3) * LDP + 2 * krp] = packbf2(pb[it][0].w, pb[it][1].w);
    }
  };

  const int NK = NH / BK;
  loadA(0); loadB(0);
  storeA(0); storeB(0);
  __syncthreads();

  for (int i = 0; i < NK; ++i) {
    const int cur = i & 1;
    if (i + 1 < NK) { loadA((i + 1) * BK); loadB((i + 1) * BK); }

    Frag a[4], b[2];
    #pragma unroll
    for (int im = 0; im < 4; ++im) {
      const unsigned short* p = &ldsA[cur][(wm * 64 + im * 16 + r) * LDP + half * 8];
      a[im].h[0] = *(const bf16x8*)(p);
      a[im].h[1] = *(const bf16x8*)(p + 16);
    }
    #pragma unroll
    for (int in = 0; in < 2; ++in) {
      const unsigned short* p = &ldsB[cur][(wn * 32 + in * 16 + r) * LDP + half * 8];
      b[in].h[0] = *(const bf16x8*)(p);
      b[in].h[1] = *(const bf16x8*)(p + 16);
    }
    #pragma unroll
    for (int im = 0; im < 4; ++im)
      #pragma unroll
      for (int in = 0; in < 2; ++in)
        acc[im][in] = __builtin_amdgcn_wmma_f32_16x16x32_bf16(
            false, a[im].v, false, b[in].v, (short)0, acc[im][in], false, false);

    if (i + 1 < NK) { storeA(cur ^ 1); storeB(cur ^ 1); }
    __syncthreads();
  }

  // epilogue: + bias, exact-erf gelu, bf16 store
  #pragma unroll
  for (int im = 0; im < 4; ++im) {
    #pragma unroll
    for (int in = 0; in < 2; ++in) {
      int gm = m0 + wm * 64 + im * 16 + half * 8;
      int gn = n0 + wn * 32 + in * 16 + r;
      unsigned short* op = hout + (size_t)(e * NC + gm) * NI + gn;
      float bv = bias[in];
      #pragma unroll
      for (int t = 0; t < 8; ++t) {
        float v = acc[im][in][t] + bv;
        float g = 0.5f * v * (1.0f + erff(v * 0.70710678118654752f));
        op[(size_t)t * NI] = f2bf(g);
      }
    }
  }
}

// ---------------------------------------------------------------- pass 2 ----
__global__ __launch_bounds__(256)
void thor_gemm2_bias(const unsigned short* __restrict__ hbuf, // [E, C, I] bf16
                     const float* __restrict__ w2,            // [E, I, H]
                     const float* __restrict__ b2,            // [E, 1, H]
                     float* __restrict__ y)                   // [E*C, H]
{
  __shared__ unsigned short ldsA[2][BM * LDP];
  __shared__ unsigned short ldsB[2][BN * LDP];

  const int e  = blockIdx.z;
  const int m0 = blockIdx.y * BM;
  const int n0 = blockIdx.x * BN;

  const int tid  = threadIdx.x;
  const int lane = tid & 31;
  const int wave = tid >> 5;
  const int wm   = wave & 1;
  const int wn   = wave >> 1;
  const int half = lane >> 4;
  const int r    = lane & 15;

  const unsigned short* he = hbuf + (size_t)(e * NC + m0) * NI;
  const float* w2e = w2 + (size_t)e * NI * NH + n0;
  const float* b2e = b2 + (size_t)e * NH + n0;

  f32x8 acc[4][2];
  for (int i = 0; i < 4; ++i)
    for (int j = 0; j < 2; ++j)
      for (int t = 0; t < 8; ++t) acc[i][j][t] = 0.0f;

  float bias[2];
  bias[0] = b2e[wn * 32 + 0 * 16 + r];
  bias[1] = b2e[wn * 32 + 1 * 16 + r];

#ifndef HAVE_ASYNC_COPY
  uint2 pa[4];
#endif
  float4 pb[2][2];

  // A tile (already bf16): 128x32 = 512 x 16B chunks, 2 per thread.
#ifdef HAVE_ASYNC_COPY
  auto asyncA = [&](int kb, int buf) {       // memory -> LDS, no VGPR data
    #pragma unroll
    for (int it = 0; it < 2; ++it) {
      int idx = tid + it * 256;
      int row = idx >> 2;                    // 4 chunks of 8 bf16 per row
      int c8  = (idx & 3) << 3;
      __builtin_amdgcn_global_load_async_to_lds_b128(
          (gv4i*)(const void*)(he + (size_t)row * NI + kb + c8),
          (lv4i*)(void*)&ldsA[buf][row * LDP + c8], 0, 0);
    }
  };
#else
  auto loadA = [&](int kb) {
    #pragma unroll
    for (int it = 0; it < 2; ++it) {
      int idx = tid + it * 256;
      int row = idx >> 2;
      int c8  = (idx & 3) << 3;
      pa[2 * it + 0] = *(const uint2*)(he + (size_t)row * NI + kb + c8);
      pa[2 * it + 1] = *(const uint2*)(he + (size_t)row * NI + kb + c8 + 4);
    }
  };
  auto storeA = [&](int buf) {
    #pragma unroll
    for (int it = 0; it < 2; ++it) {
      int idx = tid + it * 256;
      int row = idx >> 2;
      int c8  = (idx & 3) << 3;
      *(uint2*)&ldsA[buf][row * LDP + c8]     = pa[2 * it + 0];
      *(uint2*)&ldsA[buf][row * LDP + c8 + 4] = pa[2 * it + 1];
    }
  };
#endif
  auto loadB = [&](int kb) {
    #pragma unroll
    for (int it = 0; it < 2; ++it) {
      int idx = tid + it * 256;
      int krp = idx >> 5;
      int c4  = (idx & 31) << 2;
      const float* g = w2e + (size_t)(kb + 2 * krp) * NH + c4;
      pb[it][0] = *(const float4*)g;
      pb[it][1] = *(const float4*)(g + NH);
    }
  };
  auto storeB = [&](int buf) {
    #pragma unroll
    for (int it = 0; it < 2; ++it) {
      int idx = tid + it * 256;
      int krp = idx >> 5;
      int c4  = (idx & 31) << 2;
      unsigned short* base = &ldsB[buf][0];
      *(unsigned*)&base[(c4 + 0) * LDP + 2 * krp] = packbf2(pb[it][0].x, pb[it][1].x);
      *(unsigned*)&base[(c4 + 1) * LDP + 2 * krp] = packbf2(pb[it][0].y, pb[it][1].y);
      *(unsigned*)&base[(c4 + 2) * LDP + 2 * krp] = packbf2(pb[it][0].z, pb[it][1].z);
      *(unsigned*)&base[(c4 + 3) * LDP + 2 * krp] = packbf2(pb[it][0].w, pb[it][1].w);
    }
  };

  const int NK = NI / BK;
#ifdef HAVE_ASYNC_COPY
  asyncA(0, 0);
  loadB(0); storeB(0);
  wait_async0();
#else
  loadA(0); loadB(0);
  storeA(0); storeB(0);
#endif
  __syncthreads();

  for (int i = 0; i < NK; ++i) {
    const int cur = i & 1;
    if (i + 1 < NK) {
#ifdef HAVE_ASYNC_COPY
      asyncA((i + 1) * BK, cur ^ 1);
#else
      loadA((i + 1) * BK);
#endif
      loadB((i + 1) * BK);
    }

    Frag a[4], b[2];
    #pragma unroll
    for (int im = 0; im < 4; ++im) {
      const unsigned short* p = &ldsA[cur][(wm * 64 + im * 16 + r) * LDP + half * 8];
      a[im].h[0] = *(const bf16x8*)(p);
      a[im].h[1] = *(const bf16x8*)(p + 16);
    }
    #pragma unroll
    for (int in = 0; in < 2; ++in) {
      const unsigned short* p = &ldsB[cur][(wn * 32 + in * 16 + r) * LDP + half * 8];
      b[in].h[0] = *(const bf16x8*)(p);
      b[in].h[1] = *(const bf16x8*)(p + 16);
    }
    #pragma unroll
    for (int im = 0; im < 4; ++im)
      #pragma unroll
      for (int in = 0; in < 2; ++in)
        acc[im][in] = __builtin_amdgcn_wmma_f32_16x16x32_bf16(
            false, a[im].v, false, b[in].v, (short)0, acc[im][in], false, false);

    if (i + 1 < NK) {
#ifndef HAVE_ASYNC_COPY
      storeA(cur ^ 1);
#endif
      storeB(cur ^ 1);
    }
#ifdef HAVE_ASYNC_COPY
    wait_async0();     // async writes to LDS must land before the barrier
#endif
    __syncthreads();
  }

  #pragma unroll
  for (int im = 0; im < 4; ++im) {
    #pragma unroll
    for (int in = 0; in < 2; ++in) {
      int gm = m0 + wm * 64 + im * 16 + half * 8;
      int gn = n0 + wn * 32 + in * 16 + r;
      float* op = y + (size_t)(e * NC + gm) * NH + gn;
      float bv = bias[in];
      #pragma unroll
      for (int t = 0; t < 8; ++t)
        op[(size_t)t * NH] = acc[im][in][t] + bv;
    }
  }
}

// ---------------------------------------------------------------- launch ----
extern "C" void kernel_launch(void* const* d_in, const int* in_sizes, int n_in,
                              void* d_out, int out_size, void* d_ws, size_t ws_size,
                              hipStream_t stream) {
  const float* x  = (const float*)d_in[0];   // [E*C, H]
  const float* w1 = (const float*)d_in[1];   // [E, H, I]
  const float* b1 = (const float*)d_in[2];   // [E, 1, I]
  const float* w2 = (const float*)d_in[3];   // [E, I, H]
  const float* b2 = (const float*)d_in[4];   // [E, 1, H]
  float* out = (float*)d_out;                // [E*C, H]
  unsigned short* hbuf = (unsigned short*)d_ws; // bf16 [E, C, I], 128 MiB

  dim3 blk(256, 1, 1);
  dim3 g1(NI / BN, NC / BM, NE);   // N fastest -> expert locality in L2
  hipLaunchKernelGGL(thor_gemm1_gelu, g1, blk, 0, stream, x, w1, b1, hbuf);
  dim3 g2(NH / BN, NC / BM, NE);
  hipLaunchKernelGGL(thor_gemm2_bias, g2, blk, 0, stream, hbuf, w2, b2, out);
}
